// WeTe_78434692759818
// MI455X (gfx1250) — matrix operand order repaired and checked by reading, not compiled, not run on hardware
//
#include <hip/hip_runtime.h>

// ---------------- CDNA5 WMMA types ----------------
typedef __attribute__((ext_vector_type(16))) __bf16          v16bf;
typedef __attribute__((ext_vector_type(16))) unsigned short  v16us;
typedef __attribute__((ext_vector_type(8)))  unsigned short  v8us;
typedef __attribute__((ext_vector_type(8)))  float           v8f;

// ---------------- problem dims ----------------
#define TV   30000   // vocab
#define TVP  30016   // vocab padded to mult of 32 (938 K-steps)
#define TK   100     // topics
#define TKP  128     // topics padded
#define THH  300     // topic embed dim
#define THP  320     // padded
#define TDH  256     // hidden dim
#define TB   512     // batch (docs)
#define TL   1024    // tokens per doc
#define NS   5       // Weibull samples
#define N3P  208     // 2K padded to mult of 16
#define KSPL 8       // K-split factor for GEMM1

// ---------------- helpers ----------------
__device__ inline unsigned short f2bf(float f) {
  __bf16 h = (__bf16)f;                    // native cvt bf16
  return __builtin_bit_cast(unsigned short, h);
}

__device__ inline float wave_sum(float v) {
#pragma unroll
  for (int m = 16; m >= 1; m >>= 1) v += __shfl_xor(v, m, 32);
  return v;
}
__device__ inline float wave_max(float v) {
#pragma unroll
  for (int m = 16; m >= 1; m >>= 1) v = fmaxf(v, __shfl_xor(v, m, 32));
  return v;
}

// Fragment layouts per CDNA5 ISA 7.12.2 (wave32):
//   A 16x32 : lane m=ln&15, half h=ln>>4, elem e -> k = (e>>3)*16 + h*8 + (e&7)
//   B 32x16 : lane n=ln&15, half h=ln>>4, elem e -> k = (e>>3)*16 + h*8 + (e&7)
//   C 16x16 : elem j -> row m = h*8 + j, col n = ln&15
__device__ inline v16bf ldA(const unsigned short* __restrict__ ap) { // ap = row + k0 + h*8
  v8us a0 = *(const v8us*)ap;
  v8us a1 = *(const v8us*)(ap + 16);
  v16us au = __builtin_shufflevector(a0, a1, 0, 1, 2, 3, 4, 5, 6, 7,
                                     8, 9, 10, 11, 12, 13, 14, 15);
  return __builtin_bit_cast(v16bf, au);
}
__device__ inline v16bf ldB(const unsigned short* __restrict__ bp) { // bp = row + k0 + h*8
  v8us b0 = *(const v8us*)bp;
  v8us b1 = *(const v8us*)(bp + 16);
  v16us bu = __builtin_shufflevector(b0, b1, 0, 1, 2, 3, 4, 5, 6, 7,
                                     8, 9, 10, 11, 12, 13, 14, 15);
  return __builtin_bit_cast(v16bf, bu);
}
__device__ inline v8f wmma_bf16(v16bf a, v16bf b, v8f c) {
  return __builtin_amdgcn_wmma_f32_16x16x32_bf16(false, a, false, b,
                                                 (short)0, c, false, false);
}

// ---------------- JAX threefry2x32, key = PRNGKey(42) = (0,42) ----------------
__device__ inline unsigned rotl32(unsigned x, int r) { return (x << r) | (x >> (32 - r)); }

__device__ inline float jax_uniform(unsigned idx) {
  const unsigned n    = NS * TB * TK;   // 256000 (even)
  const unsigned half = n >> 1;         // 128000
  unsigned j  = (idx < half) ? idx : (idx - half);
  unsigned x0 = j, x1 = j + half;       // counters: first/second halves of iota
  const unsigned ks0 = 0u, ks1 = 42u;
  const unsigned ks2 = 0x1BD11BDAu ^ ks0 ^ ks1;
  const unsigned ks[3] = {ks0, ks1, ks2};
  x0 += ks0; x1 += ks1;
  const int R0[4] = {13, 15, 26, 6};
  const int R1[4] = {17, 29, 16, 24};
#pragma unroll
  for (int i = 0; i < 5; ++i) {
#pragma unroll
    for (int r = 0; r < 4; ++r) {
      int rot = (i & 1) ? R1[r] : R0[r];
      x0 += x1; x1 = rotl32(x1, rot); x1 ^= x0;
    }
    x0 += ks[(i + 1) % 3];
    x1 += ks[(i + 2) % 3] + (unsigned)(i + 1);
  }
  unsigned o = (idx < half) ? x0 : x1;
  unsigned f = (o >> 9) | 0x3F800000u;            // [1,2)
  return __builtin_bit_cast(float, f) - 1.0f;     // [0,1)
}

__device__ inline float softplusf(float v) {
  return (v > 20.f) ? v : log1pf(expf(v));
}

// ---------------- conversion / packing kernels ----------------

__global__ void k_init(float* acc) {
  if (threadIdx.x < 3 && blockIdx.x == 0) acc[threadIdx.x] = 0.f;
}

__global__ void c_bow(const float* __restrict__ in, unsigned short* __restrict__ o) {
  int i = blockIdx.x * blockDim.x + threadIdx.x;
  if (i >= TB * TVP) return;
  int b = i / TVP, k = i % TVP;
  o[i] = (k < TV) ? f2bf(in[(size_t)b * TV + k]) : (unsigned short)0;
}
__global__ void c_w1t(const float* __restrict__ in, unsigned short* __restrict__ o) {
  int i = blockIdx.x * blockDim.x + threadIdx.x;
  if (i >= TDH * TVP) return;
  int n = i / TVP, k = i % TVP;
  o[i] = (k < TV) ? f2bf(in[(size_t)k * TDH + n]) : (unsigned short)0;
}
__global__ void c_w2t(const float* __restrict__ in, unsigned short* __restrict__ o) {
  int i = blockIdx.x * blockDim.x + threadIdx.x;
  if (i >= TDH * TDH) return;
  int n = i / TDH, k = i % TDH;
  o[i] = f2bf(in[(size_t)k * TDH + n]);
}
__global__ void c_w3t(const float* __restrict__ in, unsigned short* __restrict__ o) {
  int i = blockIdx.x * blockDim.x + threadIdx.x;
  if (i >= N3P * TDH) return;
  int n = i / TDH, k = i % TDH;
  o[i] = (n < 2 * TK) ? f2bf(in[(size_t)k * (2 * TK) + n]) : (unsigned short)0;
}
__global__ void c_rho(const float* __restrict__ in, unsigned short* __restrict__ o) {
  int i = blockIdx.x * blockDim.x + threadIdx.x;
  if (i >= TV * THP) return;
  int v = i / THP, k = i % THP;
  o[i] = (k < THH) ? f2bf(in[(size_t)v * THH + k]) : (unsigned short)0;
}
__global__ void c_alpha(const float* __restrict__ in, unsigned short* __restrict__ o) {
  int i = blockIdx.x * blockDim.x + threadIdx.x;
  if (i >= 112 * THP) return;
  int n = i / THP, k = i % THP;
  o[i] = (n < TK && k < THH) ? f2bf(in[(size_t)n * THH + k]) : (unsigned short)0;
}

// ---------------- GEMM kernels ----------------

// GEMM1 partials: bow @ W1, 2x2 register blocking + 8-way K-split.
// wave -> (kslice, 32x32 output block). 1024 waves. ppart[ks][B][DH] f32.
__global__ void k_mlp1(const unsigned short* __restrict__ bowb,
                       const unsigned short* __restrict__ w1t,
                       float* __restrict__ ppart) {
  int wave = (blockIdx.x * blockDim.x + threadIdx.x) >> 5;
  if (wave >= KSPL * 16 * 8) return;
  int ks  = wave >> 7;          // 0..7
  int rem = wave & 127;
  int m0  = (rem >> 3) * 32;    // 16 M-pairs
  int n0  = (rem & 7) * 32;     // 8  N-pairs
  // 938 steps total: slices 0..1 get 118, 2..7 get 117
  int s0 = ks * 117 + (ks < 2 ? ks : 2);
  int s1 = s0 + 117 + (ks < 2 ? 1 : 0);
  const int ln = threadIdx.x & 31;
  const int mA = ln & 15, hh = ln >> 4, nB = ln & 15;
  const unsigned short* a0r = bowb + (size_t)(m0 + mA) * TVP;
  const unsigned short* a1r = a0r + (size_t)16 * TVP;
  const unsigned short* b0r = w1t + (size_t)(n0 + nB) * TVP;
  const unsigned short* b1r = b0r + (size_t)16 * TVP;
  v8f c00 = {0.f,0.f,0.f,0.f,0.f,0.f,0.f,0.f}, c01 = c00, c10 = c00, c11 = c00;
  for (int s = s0; s < s1; ++s) {
    int k0 = s * 32;
    __builtin_prefetch(a0r + k0 + 2048, 0, 1);
    __builtin_prefetch(a1r + k0 + 2048, 0, 1);
    __builtin_prefetch(b0r + k0 + 2048, 0, 1);
    __builtin_prefetch(b1r + k0 + 2048, 0, 1);
    v16bf a0 = ldA(a0r + k0 + (hh << 3));
    v16bf a1 = ldA(a1r + k0 + (hh << 3));
    v16bf b0 = ldB(b0r + k0 + (hh << 3));
    v16bf b1 = ldB(b1r + k0 + (hh << 3));
    c00 = wmma_bf16(a0, b0, c00);
    c01 = wmma_bf16(a0, b1, c01);
    c10 = wmma_bf16(a1, b0, c10);
    c11 = wmma_bf16(a1, b1, c11);
  }
  float* pp = ppart + (size_t)ks * TB * TDH;
  int n = n0 + nB;
#pragma unroll
  for (int j = 0; j < 8; ++j) {
    int m = m0 + hh * 8 + j;
    pp[m * TDH + n]               = c00[j];
    pp[m * TDH + n + 16]          = c01[j];
    pp[(m + 16) * TDH + n]        = c10[j];
    pp[(m + 16) * TDH + n + 16]   = c11[j];
  }
}

// reduce K-split partials + bias + relu -> h1b bf16
__global__ void k_mlp1red(const float* __restrict__ ppart, const float* __restrict__ b1,
                          unsigned short* __restrict__ h1b) {
  int i = blockIdx.x * blockDim.x + threadIdx.x;
  if (i >= TB * TDH) return;
  float s = 0.f;
#pragma unroll
  for (int ks = 0; ks < KSPL; ++ks) s += ppart[(size_t)ks * TB * TDH + i];
  float v = s + b1[i & (TDH - 1)];
  h1b[i] = f2bf(v > 0.f ? v : 0.f);
}

// h2b = relu(h1 @ W2 + b2)  M=512 N=256 K=256
__global__ void k_mlp2(const unsigned short* __restrict__ h1b,
                       const unsigned short* __restrict__ w2t,
                       const float* __restrict__ b2, unsigned short* __restrict__ h2b) {
  int wave = (blockIdx.x * blockDim.x + threadIdx.x) >> 5;
  const int NT = TDH / 16;
  if (wave >= (TB / 16) * NT) return;
  int m0 = (wave / NT) * 16, n0 = (wave % NT) * 16;
  const int ln = threadIdx.x & 31;
  const int mA = ln & 15, hh = ln >> 4, nB = ln & 15;
  const unsigned short* arow = h1b + (size_t)(m0 + mA) * TDH;
  const unsigned short* brow = w2t + (size_t)(n0 + nB) * TDH;
  v8f c = {0.f,0.f,0.f,0.f,0.f,0.f,0.f,0.f};
#pragma unroll
  for (int k0 = 0; k0 < TDH; k0 += 32)
    c = wmma_bf16(ldA(arow + k0 + (hh << 3)), ldB(brow + k0 + (hh << 3)), c);
  int n = n0 + nB;
#pragma unroll
  for (int j = 0; j < 8; ++j) {
    int m = m0 + hh * 8 + j;
    float v = c[j] + b2[n];
    h2b[m * TDH + n] = f2bf(v > 0.f ? v : 0.f);
  }
}

// h3 = softplus(h2 @ W3 + b3)  M=512 N=208(pad; store n<200) K=256
__global__ void k_mlp3(const unsigned short* __restrict__ h2b,
                       const unsigned short* __restrict__ w3t,
                       const float* __restrict__ b3, float* __restrict__ h3) {
  int wave = (blockIdx.x * blockDim.x + threadIdx.x) >> 5;
  const int NT = N3P / 16;   // 13
  if (wave >= (TB / 16) * NT) return;
  int m0 = (wave / NT) * 16, n0 = (wave % NT) * 16;
  const int ln = threadIdx.x & 31;
  const int mA = ln & 15, hh = ln >> 4, nB = ln & 15;
  const unsigned short* arow = h2b + (size_t)(m0 + mA) * TDH;
  const unsigned short* brow = w3t + (size_t)(n0 + nB) * TDH;  // padded rows zero
  v8f c = {0.f,0.f,0.f,0.f,0.f,0.f,0.f,0.f};
#pragma unroll
  for (int k0 = 0; k0 < TDH; k0 += 32)
    c = wmma_bf16(ldA(arow + k0 + (hh << 3)), ldB(brow + k0 + (hh << 3)), c);
  int nn = n0 + nB;
  if (nn < 2 * TK) {
#pragma unroll
    for (int j = 0; j < 8; ++j) {
      int m = m0 + hh * 8 + j;
      h3[m * (2 * TK) + nn] = softplusf(c[j] + b3[nn]);
    }
  }
}

// theta[b,k] = mean_s clip(scale * (-log(eps+1e-10))^(1/shape)); also bf16 copy
__global__ void k_theta(const float* __restrict__ h3, float* __restrict__ theta,
                        float* __restrict__ out_theta, unsigned short* __restrict__ thetab) {
  int i = blockIdx.x * blockDim.x + threadIdx.x;
  if (i >= TB * TKP) return;
  int b = i >> 7, k = i & (TKP - 1);
  if (k >= TK) { thetab[i] = 0; return; }   // zero K-padding for WMMA
  float sh = h3[b * (2 * TK) + k];
  sh = fminf(fmaxf(sh, 0.1f), 100.f);
  float sc = h3[b * (2 * TK) + TK + k];
  sc = fminf(fmaxf(sc, 1e-4f), 1e4f);
  float inv = 1.0f / sh;
  float acc = 0.f;
#pragma unroll
  for (int s = 0; s < NS; ++s) {
    float e  = jax_uniform((unsigned)(s * TB * TK + b * TK + k));
    float th = sc * powf(-logf(e + 1e-10f), inv);
    acc += fminf(fmaxf(th, 1e-10f), 100.f);
  }
  float tv = acc * (1.0f / NS);
  theta[b * TK + k] = tv;
  out_theta[b * TK + k] = tv;
  thetab[i] = f2bf(tv);
}

// theta_norm = softmax(theta, axis=-1); one wave per doc
__global__ void k_tnorm(const float* __restrict__ theta, float* __restrict__ tn) {
  int wave = (blockIdx.x * blockDim.x + threadIdx.x) >> 5;
  if (wave >= TB) return;
  const int ln = threadIdx.x & 31;
  float v[4], mx = -1e30f;
#pragma unroll
  for (int q = 0; q < 4; ++q) {
    int k = ln + 32 * q;
    int kc = k < TK ? k : TK - 1;
    float lv = theta[wave * TK + kc];
    v[q] = (k < TK) ? lv : -1e30f;
    mx = fmaxf(mx, v[q]);
  }
  mx = wave_max(mx);
  float s = 0.f;
#pragma unroll
  for (int q = 0; q < 4; ++q) s += (v[q] > -1e29f) ? expf(v[q] - mx) : 0.f;
  s = wave_sum(s);
#pragma unroll
  for (int q = 0; q < 4; ++q) {
    int k = ln + 32 * q;
    if (k < TK) tn[wave * TK + k] = expf(v[q] - mx) / s;
  }
}

// inner_p = rho @ alpha.T, 1x7 N-blocking: one rhob A-frag feeds 7 B-frags.
// 1875 waves; rhob streamed exactly once.
__global__ void k_innerp(const unsigned short* __restrict__ rhob,
                         const unsigned short* __restrict__ alphab,
                         float* __restrict__ ip) {
  int wave = (blockIdx.x * blockDim.x + threadIdx.x) >> 5;
  if (wave >= TV / 16) return;
  int m0 = wave * 16;
  const int ln = threadIdx.x & 31;
  const int mA = ln & 15, hh = ln >> 4, nB = ln & 15;
  const unsigned short* arow = rhob + (size_t)(m0 + mA) * THP;
  v8f c[7];
#pragma unroll
  for (int t = 0; t < 7; ++t) c[t] = v8f{0.f,0.f,0.f,0.f,0.f,0.f,0.f,0.f};
#pragma unroll
  for (int k0 = 0; k0 < THP; k0 += 32) {
    v16bf a = ldA(arow + k0 + (hh << 3));
#pragma unroll
    for (int t = 0; t < 7; ++t) {
      v16bf b = ldB(alphab + (size_t)(16 * t + nB) * THP + k0 + (hh << 3));
      c[t] = wmma_bf16(a, b, c[t]);
    }
  }
#pragma unroll
  for (int t = 0; t < 7; ++t) {
    int nn = 16 * t + nB;
    if (nn < TK) {
#pragma unroll
      for (int j = 0; j < 8; ++j) ip[(size_t)(m0 + hh * 8 + j) * TK + nn] = c[t][j];
    }
  }
}

// per-vocab-row: phib = bf16 softmax(ip) (K-padded), dis/cost = clip(exp(+-ip))
__global__ void k_rowsoft(const float* __restrict__ ip, unsigned short* __restrict__ phib,
                          float* __restrict__ dis, float* __restrict__ cost) {
  int wave = (blockIdx.x * blockDim.x + threadIdx.x) >> 5;
  if (wave >= TV) return;
  const int ln = threadIdx.x & 31;
  const float* row = ip + (size_t)wave * TK;
  float v[4], mx = -1e30f;
#pragma unroll
  for (int q = 0; q < 4; ++q) {
    int k = ln + 32 * q;
    int kc = k < TK ? k : TK - 1;
    float lv = row[kc];
    v[q] = (k < TK) ? lv : -1e30f;
    mx = fmaxf(mx, v[q]);
  }
  mx = wave_max(mx);
  float s = 0.f;
#pragma unroll
  for (int q = 0; q < 4; ++q) s += (v[q] > -1e29f) ? expf(v[q] - mx) : 0.f;
  s = wave_sum(s);
#pragma unroll
  for (int q = 0; q < 4; ++q) {
    int k = ln + 32 * q;   // covers exactly 0..127 == TKP
    float p = expf(v[q] - mx) / s;
    phib[(size_t)wave * TKP + k] = (k < TK) ? f2bf(p) : (unsigned short)0;
    if (k < TK) {
      size_t o = (size_t)wave * TK + k;
      dis[o]  = fminf(fmaxf(expf(v[q]),  1e-30f), 1e10f);
      cost[o] = fminf(fmaxf(expf(-v[q]), 1e-30f), 1e10f);
    }
  }
}

// forward + backward OT costs; one wave per doc, per-lane per-topic accumulators
__global__ void k_tokens(const int* __restrict__ x, const float* __restrict__ dis,
                         const float* __restrict__ cost, const float* __restrict__ tn,
                         float* __restrict__ acc) {
  int wave = (blockIdx.x * blockDim.x + threadIdx.x) >> 5;
  if (wave >= TB) return;
  const int ln = threadIdx.x & 31;
  const int b = wave;
  int   kc[4];
  float msk[4], t[4];
  float num[4] = {0.f,0.f,0.f,0.f}, den[4] = {0.f,0.f,0.f,0.f};
#pragma unroll
  for (int q = 0; q < 4; ++q) {
    int k = ln + 32 * q;
    kc[q]  = k < TK ? k : TK - 1;
    msk[q] = k < TK ? 1.f : 0.f;
    t[q]   = tn[b * TK + kc[q]] * msk[q];
  }
  float facc = 0.f;
  for (int l = 0; l < TL; ++l) {
    int v = x[b * TL + l];
    const float* dr = dis  + (size_t)v * TK;
    const float* cr = cost + (size_t)v * TK;
    float s1 = 0.f, s2 = 0.f;
#pragma unroll
    for (int q = 0; q < 4; ++q) {
      float d  = dr[kc[q]] * msk[q];     // masked: 0 for padded topics
      float cc = cr[kc[q]];
      float w  = d * t[q];
      s1 += w;
      s2 += cc * w;
      num[q] += cc * d;
      den[q] += d;
    }
    s1 = wave_sum(s1);
    s2 = wave_sum(s2);
    facc += s2 / (s1 + 1e-30f);
  }
  if (ln == 0) atomicAdd(&acc[0], facc / (float)TL);   // forward: per-doc mean
  float bacc = 0.f;
#pragma unroll
  for (int q = 0; q < 4; ++q)
    bacc += num[q] / (den[q] + 1e-30f) * t[q];         // padded lanes contribute 0
  bacc = wave_sum(bacc);
  if (ln == 0) atomicAdd(&acc[1], bacc);               // backward
}

// TM cost: re_x = theta @ phi.T fused with Poisson LL.  4x1 M-blocking:
// one phib B-frag feeds 4 M-tiles (thetab is L2-resident). 15000 waves.
__global__ void k_tm(const unsigned short* __restrict__ thetab,
                     const unsigned short* __restrict__ phib,
                     const float* __restrict__ bow, float* __restrict__ acc) {
  int wave = (blockIdx.x * blockDim.x + threadIdx.x) >> 5;
  const int NT = TV / 16;   // 1875
  if (wave >= (TB / 64) * NT) return;
  int mg = wave / NT;            // 0..7 -> m0 = mg*64
  int n0 = (wave % NT) * 16;
  int m0 = mg * 64;
  const int ln = threadIdx.x & 31;
  const int mA = ln & 15, hh = ln >> 4, nB = ln & 15;
  const unsigned short* ar = thetab + (size_t)(m0 + mA) * TKP;
  const unsigned short* brow = phib + (size_t)(n0 + nB) * TKP;
  v8f c[4];
#pragma unroll
  for (int i = 0; i < 4; ++i) c[i] = v8f{0.f,0.f,0.f,0.f,0.f,0.f,0.f,0.f};
#pragma unroll
  for (int k0 = 0; k0 < TKP; k0 += 32) {
    v16bf b = ldB(brow + k0 + (hh << 3));
#pragma unroll
    for (int i = 0; i < 4; ++i) {
      v16bf a = ldA(ar + (size_t)16 * i * TKP + k0 + (hh << 3));
      c[i] = wmma_bf16(a, b, c[i]);
    }
  }
  int n = n0 + nB;
  float a = 0.f;
#pragma unroll
  for (int i = 0; i < 4; ++i) {
#pragma unroll
    for (int j = 0; j < 8; ++j) {
      int m = m0 + 16 * i + hh * 8 + j;
      float re = c[i][j];
      float bw = bow[(size_t)m * TV + n];
      a += bw * logf(re + 1e-10f) - re - lgammaf(bw + 1.0f);
    }
  }
  a = wave_sum(a);
  if (ln == 0) atomicAdd(&acc[2], a);
}

__global__ void k_final(const float* __restrict__ acc, float* __restrict__ out) {
  if (threadIdx.x == 0 && blockIdx.x == 0) {
    float fwd = acc[0];
    float bwd = acc[1];
    float tm  = -acc[2] / (float)TB;
    out[0] = 0.5f * fwd + 0.5f * bwd + 1.0f * tm;  // BETA=0.5, EPSILON=1.0
    out[1] = fwd;
    out[2] = bwd;
    out[3] = tm;
  }
}

// ---------------- launch ----------------
extern "C" void kernel_launch(void* const* d_in, const int* in_sizes, int n_in,
                              void* d_out, int out_size, void* d_ws, size_t ws_size,
                              hipStream_t stream) {
  (void)in_sizes; (void)n_in; (void)out_size; (void)ws_size;
  const int*   x     = (const int*)d_in[0];
  const float* bow   = (const float*)d_in[1];
  const float* rho   = (const float*)d_in[2];
  const float* alpha = (const float*)d_in[3];
  const float* W1    = (const float*)d_in[4];
  const float* b1    = (const float*)d_in[5];
  const float* W2    = (const float*)d_in[6];
  const float* b2    = (const float*)d_in[7];
  const float* W3    = (const float*)d_in[8];
  const float* b3    = (const float*)d_in[9];
  float* out  = (float*)d_out;
  char*  base = (char*)d_ws;

  // ---- shared region S (phase1 aliases phase2; stream order keeps it safe) ----
  // phase1: bowb [512][30016] bf16 (30,736,384 B) + w1t [256][30016] bf16 (15,368,192 B)
  unsigned short* bowb   = (unsigned short*)(base + 0);
  unsigned short* w1t    = (unsigned short*)(base + 30736384);
  // phase2 (after k_mlp1): rhob [30000][320] bf16 + alphab [112][320] bf16 + ip f32
  unsigned short* rhob   = (unsigned short*)(base + 0);           // 19,200,000 B
  unsigned short* alphab = (unsigned short*)(base + 19200000);    //     71,680 B
  float*          ip     = (float*)(base + 19271680);             // 12,000,000 B
  const size_t S_END = 46104576;                                  // 46.1 MB

  // ---- persistent region ----
  size_t p = S_END;
  float*          ppart  = (float*)(base + p);          p += (size_t)KSPL * TB * TDH * 4; // 4 MB
  unsigned short* w2t    = (unsigned short*)(base + p); p += (size_t)TDH * TDH * 2;
  unsigned short* w3t    = (unsigned short*)(base + p); p += (size_t)N3P * TDH * 2;
  unsigned short* h1b    = (unsigned short*)(base + p); p += (size_t)TB * TDH * 2;
  unsigned short* h2b    = (unsigned short*)(base + p); p += (size_t)TB * TDH * 2;
  float*          h3     = (float*)(base + p);          p += (size_t)TB * 2 * TK * 4;
  float*          theta  = (float*)(base + p);          p += (size_t)TB * TK * 4;
  unsigned short* thetab = (unsigned short*)(base + p); p += (size_t)TB * TKP * 2;
  float*          tn     = (float*)(base + p);          p += (size_t)TB * TK * 4;
  unsigned short* phib   = (unsigned short*)(base + p); p += (size_t)TV * TKP * 2;
  float*          dis    = (float*)(base + p);          p += (size_t)TV * TK * 4;
  float*          cost   = (float*)(base + p);          p += (size_t)TV * TK * 4;
  float*          acc    = (float*)(base + p);          // 3 floats

  const int BLK = 128;  // 4 waves/block
  auto wb = [](int waves) { return (waves + 3) / 4; };
  auto eb = [](long n) { return (int)((n + 255) / 256); };

  k_init<<<1, 32, 0, stream>>>(acc);
  // phase 1: inference MLP
  c_bow<<<eb((long)TB * TVP), 256, 0, stream>>>(bow, bowb);
  c_w1t<<<eb((long)TDH * TVP), 256, 0, stream>>>(W1, w1t);
  c_w2t<<<eb((long)TDH * TDH), 256, 0, stream>>>(W2, w2t);
  c_w3t<<<eb((long)N3P * TDH), 256, 0, stream>>>(W3, w3t);
  k_mlp1<<<wb(KSPL * 16 * 8), BLK, 0, stream>>>(bowb, w1t, ppart);
  k_mlp1red<<<eb((long)TB * TDH), 256, 0, stream>>>(ppart, b1, h1b);
  k_mlp2<<<wb((TB / 16) * (TDH / 16)), BLK, 0, stream>>>(h1b, w2t, b2, h2b);
  k_mlp3<<<wb((TB / 16) * (N3P / 16)), BLK, 0, stream>>>(h2b, w3t, b3, h3);
  k_theta<<<eb((long)TB * TKP), 256, 0, stream>>>(h3, theta, out + 4, thetab);
  k_tnorm<<<wb(TB), BLK, 0, stream>>>(theta, tn);
  // phase 2: topic tables (reuses S region, after k_mlp1 is done with it)
  c_rho<<<eb((long)TV * THP), 256, 0, stream>>>(rho, rhob);
  c_alpha<<<eb((long)112 * THP), 256, 0, stream>>>(alpha, alphab);
  k_innerp<<<wb(TV / 16), BLK, 0, stream>>>(rhob, alphab, ip);
  k_rowsoft<<<wb(TV), BLK, 0, stream>>>(ip, phib, dis, cost);
  // phase 3: losses
  k_tokens<<<wb(TB), BLK, 0, stream>>>(x, dis, cost, tn, acc);
  k_tm<<<wb((TB / 64) * (TV / 16)), BLK, 0, stream>>>(thetab, phib, bow, acc);
  k_final<<<1, 32, 0, stream>>>(acc, out);
}
